// QuantLinear_72086731096554
// MI455X (gfx1250) — matrix operand order repaired and checked by reading, not compiled, not run on hardware
//
#include <hip/hip_runtime.h>
#include <hip/hip_bf16.h>

typedef __attribute__((ext_vector_type(16))) _Float16 v16h;
typedef __attribute__((ext_vector_type(8)))  _Float16 v8h;
typedef __attribute__((ext_vector_type(2)))  _Float16 h2;
typedef __attribute__((ext_vector_type(8)))  float    v8f;
typedef __attribute__((ext_vector_type(4)))  float    v4f;

#define K_DIM   4096
#define N_DIM   11008
#define NPACK   (N_DIM / 8)     // qzeros row length in int32
#define GROUP   128
#define BM      128
#define BN      128
#define BK      32
#define LDA     40              // halves per LDS row (32 + 8 pad -> 80B stride)

// Build 16 contiguous K-values of one column as EXACT f16 integers (w - z).
// nibble w OR'd into 0x6400 is the exact f16 value (1024 + w); one
// v_pk_add_f16 with -(1024+z) (exact) gives w-z with zero rounding error.
// Per f16 pair: 1 v_perm_b32 + 1 v_or_b32 + 1 v_pk_add_f16.
__device__ __forceinline__ v16h build16(int q0, int q1, h2 nb) {
  const unsigned qe0 = (unsigned)q0 & 0x0F0F0F0Fu;
  const unsigned qo0 = ((unsigned)q0 >> 4) & 0x0F0F0F0Fu;
  const unsigned qe1 = (unsigned)q1 & 0x0F0F0F0Fu;
  const unsigned qo1 = ((unsigned)q1 >> 4) & 0x0F0F0F0Fu;
  v16h r;
#pragma unroll
  for (int j = 0; j < 4; ++j) {
    // dst byte0 = qe byte j (w_{2j}), byte2 = qo byte j (w_{2j+1}), rest 0
    const unsigned sel = (0x0Cu << 24) | ((unsigned)(4 + j) << 16) |
                         (0x0Cu << 8) | (unsigned)j;
    const unsigned t0 = __builtin_amdgcn_perm(qo0, qe0, sel) | 0x64006400u;
    const unsigned t1 = __builtin_amdgcn_perm(qo1, qe1, sel) | 0x64006400u;
    const h2 p0 = __builtin_bit_cast(h2, t0) + nb;   // v_pk_add_f16
    const h2 p1 = __builtin_bit_cast(h2, t1) + nb;
    r[2 * j]     = p0[0];  r[2 * j + 1] = p0[1];     // K = 2j, 2j+1
    r[2 * j + 8] = p1[0];  r[2 * j + 9] = p1[1];     // K = 8+2j, 8+2j+1
  }
  return r;
}

__global__ __launch_bounds__(256)
void qgemm_wmma_f16(const float* __restrict__ x,
                    const int*   __restrict__ qweight,
                    const float* __restrict__ scales,
                    const int*   __restrict__ qzeros,
                    const float* __restrict__ bias,
                    float* __restrict__ out)
{
  __shared__ __align__(16) _Float16 aTile[2][BM * LDA];

  const int tid  = threadIdx.x;
  const int lane = tid & 31;
  const int wave = tid >> 5;
  const int wm   = wave & 1;    // wave M index: 0..1  -> 64 rows each
  const int wn   = wave >> 1;   // wave N index: 0..3  -> 32 cols each
  const int l16  = lane & 15;
  const int lHi  = lane >> 4;   // 0/1: K-half select per WMMA layout

  const int mBlk = blockIdx.y * BM;
  const int nBlk = blockIdx.x * BN;

  // --- A staging: 256 threads cover 128 rows x 32 K of f32 (16 floats each)
  const int arow = tid >> 1;
  const int akk  = (tid & 1) * 16;
  const float* aPtr = x + (size_t)(mBlk + arow) * K_DIM + akk;
  _Float16* aStage0 = &aTile[0][arow * LDA + akk];
  _Float16* aStage1 = &aTile[1][arow * LDA + akk];

  // --- this wave's two output column subtiles (one column per lane each)
  const int n0 = nBlk + wn * 32 + l16;
  const int n1 = n0 + 16;

  v8f acc[4][2] = {};
  const v8f vzero = {};

  // --- prologue: A tile 0 -> LDS buf 0, and B dwords for K-step 0 -> regs
  v4f aReg[4];
#pragma unroll
  for (int i = 0; i < 4; ++i) aReg[i] = ((const v4f*)aPtr)[i];

  int qn0, qn1, qn2, qn3;     // pipelined B dwords (next K-step)
  {
    const int* qp = qweight + (size_t)(lHi * 2) * N_DIM;
    qn0 = qp[n0];  qn1 = qp[N_DIM + n0];
    qn2 = qp[n1];  qn3 = qp[N_DIM + n1];
  }

  {
    v8h h0, h1;
#pragma unroll
    for (int j = 0; j < 4; ++j) {
      h0[j]     = (_Float16)aReg[0][j];  h0[j + 4] = (_Float16)aReg[1][j];
      h1[j]     = (_Float16)aReg[2][j];  h1[j + 4] = (_Float16)aReg[3][j];
    }
    *(v8h*)aStage0       = h0;
    *(v8h*)(aStage0 + 8) = h1;
  }
  __syncthreads();

  int buf = 0;
  for (int g = 0; g < K_DIM / GROUP; ++g) {
    // group constants: f32 scale per column, exact f16 -(1024+z) per column
    const float s0  = scales[g * N_DIM + n0];
    const float s1  = scales[g * N_DIM + n1];
    const int   zq0 = qzeros[g * NPACK + (n0 >> 3)];
    const int   zq1 = qzeros[g * NPACK + (n1 >> 3)];
    const float z0  = (float)((zq0 >> ((n0 & 7) * 4)) & 0xF);
    const float z1  = (float)((zq1 >> ((n1 & 7) * 4)) & 0xF);
    const _Float16 nb0h = (_Float16)(-(1024.0f + z0));   // exact in f16
    const _Float16 nb1h = (_Float16)(-(1024.0f + z1));
    const h2 nb0 = {nb0h, nb0h};
    const h2 nb1 = {nb1h, nb1h};

    v8f accg[4][2];   // per-group integer-weight accumulators

#pragma unroll
    for (int kk = 0; kk < GROUP / BK; ++kk) {
      const int k0 = g * GROUP + kk * BK;
      const bool more = (k0 + BK) < K_DIM;

      // consume pipelined B dwords, immediately refill for next K-step
      const int q00 = qn0, q01 = qn1, q10 = qn2, q11 = qn3;
      if (more) {
        const int* qp = qweight +
            (size_t)(((k0 + BK) >> 3) + lHi * 2) * N_DIM;
        qn0 = qp[n0];  qn1 = qp[N_DIM + n0];
        qn2 = qp[n1];  qn3 = qp[N_DIM + n1];
        __builtin_prefetch(qp + 4 * N_DIM + n0, 0, 1);  // K-step after next
      }

      // prefetch next A tile (global f32 -> regs), overlapped with compute
      if (more) {
        const float* ap = aPtr + k0 + BK;
#pragma unroll
        for (int i = 0; i < 4; ++i) aReg[i] = ((const v4f*)ap)[i];
      }

      const v16h bf0 = build16(q00, q01, nb0);
      const v16h bf1 = build16(q10, q11, nb1);

      // A fragments from LDS: rows wm*64+ms*16+l16, two 16B runs per lane
      const _Float16* abase = &aTile[buf][(wm * 64 + l16) * LDA + lHi * 16];
      v16h af[4];
#pragma unroll
      for (int ms = 0; ms < 4; ++ms) {
        const _Float16* ap = abase + ms * 16 * LDA;
        ((v8h*)&af[ms])[0] = *(const v8h*)ap;
        ((v8h*)&af[ms])[1] = *(const v8h*)(ap + 8);
      }

#pragma unroll
      for (int ms = 0; ms < 4; ++ms) {
        if (kk == 0) {   // first K-step of group: C = inline 0 (no zero movs)
          accg[ms][0] = __builtin_amdgcn_wmma_f32_16x16x32_f16(
              false, af[ms], false, bf0, (short)0, vzero, false, false);
          accg[ms][1] = __builtin_amdgcn_wmma_f32_16x16x32_f16(
              false, af[ms], false, bf1, (short)0, vzero, false, false);
        } else {
          accg[ms][0] = __builtin_amdgcn_wmma_f32_16x16x32_f16(
              false, af[ms], false, bf0, (short)0, accg[ms][0], false, false);
          accg[ms][1] = __builtin_amdgcn_wmma_f32_16x16x32_f16(
              false, af[ms], false, bf1, (short)0, accg[ms][1], false, false);
        }
      }

      // Single barrier per K-step is sufficient: this iteration's LDS reads
      // are consumed (dscnt-waited) before the barrier, and the stores below
      // target the *other* buffer, whose last readers synchronized at the
      // previous iteration's barrier.
      if (more) {
        _Float16* dst = buf ? aStage0 : aStage1;
        v8h h0, h1;
#pragma unroll
        for (int j = 0; j < 4; ++j) {
          h0[j]     = (_Float16)aReg[0][j];  h0[j + 4] = (_Float16)aReg[1][j];
          h1[j]     = (_Float16)aReg[2][j];  h1[j + 4] = (_Float16)aReg[3][j];
        }
        *(v8h*)dst       = h0;
        *(v8h*)(dst + 8) = h1;
      }
      __syncthreads();
      buf ^= 1;
    }

    // fold the group's exact integer accumulation with its f32 scale
#pragma unroll
    for (int ms = 0; ms < 4; ++ms) {
      acc[ms][0] = acc[ms][0] + accg[ms][0] * s0;   // v_pk_fma_f32 x4
      acc[ms][1] = acc[ms][1] + accg[ms][1] * s1;
    }
  }

  // --- epilogue: bias add + store.  C layout: VGPR v, lanes 0-15 -> M=v,
  // lanes 16-31 -> M=v+8; column = l16 within the subtile.
  const int rowBase = mBlk + wm * 64 + lHi * 8;
#pragma unroll
  for (int ns = 0; ns < 2; ++ns) {
    const int col = nBlk + wn * 32 + ns * 16 + l16;
    const float b = bias[col];
#pragma unroll
    for (int ms = 0; ms < 4; ++ms) {
#pragma unroll
      for (int v = 0; v < 8; ++v) {
        const int row = rowBase + ms * 16 + v;
        out[(size_t)row * N_DIM + col] = acc[ms][ns][v] + b;
      }
    }
  }
}

extern "C" void kernel_launch(void* const* d_in, const int* in_sizes, int n_in,
                              void* d_out, int out_size, void* d_ws, size_t ws_size,
                              hipStream_t stream) {
  (void)n_in; (void)out_size; (void)d_ws; (void)ws_size;
  const float* x    = (const float*)d_in[0];
  const int*   qw   = (const int*)d_in[1];
  const float* sc   = (const float*)d_in[2];
  const int*   qz   = (const int*)d_in[3];
  const float* bias = (const float*)d_in[4];
  float*       out  = (float*)d_out;

  const int M = in_sizes[0] / K_DIM;        // 2*2048 = 4096 rows
  dim3 grid(N_DIM / BN, M / BM);            // 86 x 32
  dim3 block(256);                          // 8 waves (wave32)
  qgemm_wmma_f16<<<grid, block, 0, stream>>>(x, qw, sc, qz, bias, out);
}
